// UserEncoder_28767690949343
// MI455X (gfx1250) — matrix operand order repaired and checked by reading, not compiled, not run on hardware
//
#include <hip/hip_runtime.h>
#include <hip/hip_bf16.h>

// ---------------------------------------------------------------------------
// Problem constants (from the reference)
// ---------------------------------------------------------------------------
#define D     300     // embedding dim
#define H     15      // heads
#define DK    20      // per-head dim
#define SS    50      // sequence length
#define SP    64      // padded sequence (4 x 16)
#define KP    320     // padded K dim: 300 data + 1 bias col + pad (10 x 32)
#define QDIM  200     // additive attention dim
#define BATCH 1024
#define ROWS  (BATCH * SS)   // 51200

typedef _Float16 f16;
typedef f16   v16h __attribute__((ext_vector_type(16)));
typedef f16   v8h  __attribute__((ext_vector_type(8)));
typedef float v8f  __attribute__((ext_vector_type(8)));

static __device__ __forceinline__ v16h cat8(v8h lo, v8h hi) {
    return __builtin_shufflevector(lo, hi, 0,1,2,3,4,5,6,7,8,9,10,11,12,13,14,15);
}

// D = A x B + C, f16 inputs, f32 accumulate (v_wmma_f32_16x16x32_f16)
static __device__ __forceinline__ v8f wmma_f16(v16h a, v16h b, v8f c) {
    return __builtin_amdgcn_wmma_f32_16x16x32_f16(
        false, a, false, b, (short)0, c, false, false);
}

// A-fragment (16x32 f16) from a row-major f16 buffer, pitch in halves.
// lanes 0-15: K = k0 + [0..15]; lanes 16-31: K = k0 + [16..31].
static __device__ __forceinline__ v16h load_a(const f16* src, int row0, int k0,
                                              int pitch, int lane) {
    int hi = lane >> 4, m = lane & 15;
    const f16* r = src + (size_t)(row0 + m) * pitch + k0 + hi * 16;
    return cat8(*(const v8h*)r, *(const v8h*)(r + 8));
}

// B-fragment (32x16 f16), src stored "N-major": src[n][k] = B[k][n].
// lane's N = n0 + (lane&15); halves j -> K = k0 + hi*16 + j (32 contiguous B).
static __device__ __forceinline__ v16h load_bT(const f16* src, int n0, int k0,
                                               int pitch, int lane) {
    int hi = lane >> 4, n = lane & 15;
    return *(const v16h*)(src + (size_t)(n0 + n) * pitch + k0 + hi * 16);
}

// -------- CDNA5 async global->LDS copy (ASYNCcnt-tracked, no VGPR bounce) ----
static __device__ __forceinline__ void async_ld_b128(unsigned lds_off,
                                                     const void* gaddr) {
    // ISA 15.18.3: GLOBAL_LOAD_ASYNC_TO_LDS_B128  (LDS address from the VDST operand)
    asm volatile("global_load_async_to_lds_b128 %0, %1, off"
                 :: "v"(lds_off), "v"(gaddr) : "memory");
}

static __device__ __forceinline__ void wait_async0() {
#if __has_builtin(__builtin_amdgcn_s_wait_asynccnt)
    __builtin_amdgcn_s_wait_asynccnt(0);
#else
    asm volatile("s_wait_asynccnt 0x0" ::: "memory");
#endif
}

// ---------------------------------------------------------------------------
// Kernel 1: pack weights into per-lane WMMA B-fragment order (f16), bias folded
// in as K-row 300.  Proj part: [mat][head][kt][nt][512]; Add part: [kt][nt][512].
// ---------------------------------------------------------------------------
__global__ __launch_bounds__(32) void pack_kernel(
    const float* __restrict__ Wq,  const float* __restrict__ Wk,
    const float* __restrict__ Wv,  const float* __restrict__ Wqp,
    const float* __restrict__ Wkp,
    const float* __restrict__ bq,  const float* __restrict__ bk,
    const float* __restrict__ bv,  const float* __restrict__ bqp,
    const float* __restrict__ bkp,
    const float* __restrict__ Wadd, const float* __restrict__ badd,
    f16* __restrict__ Wpk, f16* __restrict__ Apk)
{
    const int bid  = blockIdx.x;
    const int lane = threadIdx.x & 31;
    const int hi = lane >> 4, ln = lane & 15;

    if (bid < 5 * H * 10 * 2) {                 // 1500 projection fragment tiles
        int m   = bid / 300;
        int rem = bid % 300;
        int h   = rem / 20;
        int r2  = rem % 20;
        int kt  = r2 >> 1, nt = r2 & 1;
        const float* W    = (m == 0) ? Wq : (m == 1) ? Wk : (m == 2) ? Wv
                          : (m == 3) ? Wqp : Wkp;
        const float* bias = (m == 0) ? bq : (m == 1) ? bk : (m == 2) ? bv
                          : (m == 3) ? bqp : bkp;
        f16* dst = Wpk + (size_t)bid * 512 + lane * 16;
        int colL = nt * 16 + ln;                // column within 32-wide head slice
        int ncol = h * DK + colL;               // column in full [D,D] weight
        #pragma unroll
        for (int j = 0; j < 16; ++j) {
            int K = kt * 32 + hi * 16 + j;
            float v = 0.0f;
            if (colL < DK) {
                if (K < D)       v = W[(size_t)K * D + ncol];
                else if (K == D) v = bias[ncol];    // bias row (X col 300 == 1)
            }
            dst[j] = (f16)v;
        }
    } else {                                    // 130 additive-attn fragment tiles
        int t  = bid - 1500;
        int kt = t / 13, nt = t % 13;
        f16* dst = Apk + (size_t)t * 512 + lane * 16;
        int col = nt * 16 + ln;
        #pragma unroll
        for (int j = 0; j < 16; ++j) {
            int K = kt * 32 + hi * 16 + j;
            float v = 0.0f;
            if (col < QDIM) {
                if (K < D)       v = Wadd[(size_t)K * QDIM + col];
                else if (K == D) v = badd[col];
            }
            dst[j] = (f16)v;
        }
    }
}

// ---------------------------------------------------------------------------
// Kernel 2: embedding gather + f16 conversion into padded [ROWS, KP] buffers.
// One b128 store (8 halves) per thread.  Column 300 = constant-1 bias column.
// ---------------------------------------------------------------------------
__global__ __launch_bounds__(256) void prep_kernel(
    const float* __restrict__ uv, const float* __restrict__ emb,
    const int* __restrict__ ids,
    f16* __restrict__ Xv, f16* __restrict__ Xe)
{
    const int CPR = KP / 8;                         // 40 chunks per row
    const size_t nchunk = (size_t)ROWS * CPR;       // per buffer
    size_t i = (size_t)blockIdx.x * blockDim.x + threadIdx.x;
    if (i >= 2 * nchunk) return;
    const int which = (i >= nchunk);
    const size_t j = which ? i - nchunk : i;
    const int row = (int)(j / CPR), c8 = (int)(j % CPR);
    const float* src;
    if (!which) {
        src = uv + (size_t)row * D;
    } else {
        int id = ids[row];
        src = emb + (size_t)id * D;
    }
    v8h o;
    #pragma unroll
    for (int k = 0; k < 8; ++k) {
        int c = c8 * 8 + k;
        float v = (c < D) ? src[c] : (c == D ? 1.0f : 0.0f);
        o[k] = (f16)v;
    }
    *(v8h*)((which ? Xe : Xv) + (size_t)row * KP + c8 * 8) = o;
}

// ---------------------------------------------------------------------------
// Kernel 3: fused per-(b,h) disentangled attention.
// 256 threads (8 wave32) per block.  X staged into LDS with async-tensor
// copies, all GEMMs via v_wmma_f32_16x16x32_f16.
// LDS: 40KB X-stage (overlaid with Sc/At scratch) + 20KB Q/K/V = 60KB.
// ---------------------------------------------------------------------------
__global__ __launch_bounds__(256) void attn_kernel(
    const f16* __restrict__ Xv, const f16* __restrict__ Xe,
    const f16* __restrict__ Wpk,
    float* __restrict__ ctx)
{
    __shared__ __align__(128) union SMem {
        f16 X[SP * KP];                             // 40960 B (phase 1 staging)
        struct {
            float Sc[SP * SP];                      // 16384 B (phase 2/3)
            f16   At[SP * SP];                      //  8192 B (phase 3/4)
        } s;
    } u;
    __shared__ __align__(64) f16 Qc_s[SP * 32];
    __shared__ __align__(64) f16 Kc_s[SP * 32];
    __shared__ __align__(64) f16 Qp_s[SP * 32];
    __shared__ __align__(64) f16 Kp_s[SP * 32];
    __shared__ __align__(64) f16 VcT_s[32 * SP];    // Vc transposed [dk][s]

    const int bid = blockIdx.x;
    const int b = bid / H, h = bid % H;
    const int tid = threadIdx.x, wave = tid >> 5, lane = tid & 31;
    const int hi = lane >> 4, ln = lane & 15;
    const int mt = wave >> 1, nt = wave & 1;        // wave's output tile

    const unsigned lbase = (unsigned)(size_t)(void*)u.X;

    // ---- Phase 1: projections in two passes through the shared X stage.
    // pass 0: X = Xv -> Qc, Kc, Vc ; pass 1: X = Xe -> Qp, Kp
    for (int pass = 0; pass < 2; ++pass) {
        if (pass) __syncthreads();                  // pass-0 X reads complete
        {   // stage rows 0..49 via async global->LDS b128 copies
            const f16* src_b = (pass ? Xe : Xv) + (size_t)b * SS * KP;
            for (int c = tid; c < SS * (KP / 8); c += 256) {
                int row = c / (KP / 8), c16 = c % (KP / 8);
                async_ld_b128(lbase + (unsigned)(row * (KP * 2) + c16 * 16),
                              (const char*)(src_b + (size_t)row * KP) + c16 * 16);
            }
            if (pass == 0) {                        // zero pad rows 50..63 once
                for (int c = tid; c < (SP - SS) * (KP / 8); c += 256) {
                    int row = SS + c / (KP / 8), c16 = c % (KP / 8);
                    v8h z = {};
                    *(v8h*)(u.X + (size_t)row * KP + c16 * 8) = z;
                }
            }
            wait_async0();
        }
        __syncthreads();

        const int nmat = pass ? 2 : 3;
        const int mat0 = pass ? 3 : 0;
        v8f acc[3] = {{}, {}, {}};
        const f16* wp[3];
        #pragma unroll
        for (int i = 0; i < 3; ++i) {
            int mi = (i < nmat) ? (mat0 + i) : mat0;
            wp[i] = Wpk + ((((size_t)mi * H + h) * 10) * 2 + nt) * 512;
            __builtin_prefetch(wp[i] + (size_t)lane * 16, 0, 1);
        }
        #pragma unroll
        for (int kt = 0; kt < 10; ++kt) {
            v16h a = load_a(u.X, mt * 16, kt * 32, KP, lane);   // shared A frag
            #pragma unroll
            for (int i = 0; i < 3; ++i) {
                if (i < nmat) {
                    v16h bf = *(const v16h*)(wp[i] + (size_t)kt * 1024 + lane * 16);
                    acc[i] = wmma_f16(a, bf, acc[i]);
                }
            }
        }
        // store projection tiles
        #pragma unroll
        for (int i = 0; i < 3; ++i) {
            if (i >= nmat) continue;
            const int mat = mat0 + i;
            if (mat == 2) {                          // Vc stored transposed
                #pragma unroll
                for (int r = 0; r < 8; ++r)
                    VcT_s[(nt * 16 + ln) * SP + mt * 16 + r + hi * 8] = (f16)acc[i][r];
            } else {
                f16* dst = (mat == 0) ? Qc_s : (mat == 1) ? Kc_s
                         : (mat == 3) ? Qp_s : Kp_s;
                #pragma unroll
                for (int r = 0; r < 8; ++r)
                    dst[(mt * 16 + r + hi * 8) * 32 + nt * 16 + ln] = (f16)acc[i][r];
            }
        }
    }
    __syncthreads();   // projections visible; X stage dead -> Sc may be written

    // ---- Phase 2: scores = (Qc Kc^T + Qc Kp^T + Qp Kc^T) / sqrt(3*DK)
    const float scale = 0.12909944487358056f;       // 1/sqrt(60)
    for (int t = wave; t < 16; t += 8) {
        const int smt = t >> 2, snt = t & 3;
        v16h aC = load_a(Qc_s, smt * 16, 0, 32, lane);
        v16h aP = load_a(Qp_s, smt * 16, 0, 32, lane);
        v16h bC = load_bT(Kc_s, snt * 16, 0, 32, lane);
        v16h bP = load_bT(Kp_s, snt * 16, 0, 32, lane);
        v8f acc = {};
        acc = wmma_f16(aC, bC, acc);
        acc = wmma_f16(aC, bP, acc);
        acc = wmma_f16(aP, bC, acc);
        #pragma unroll
        for (int r = 0; r < 8; ++r)
            u.s.Sc[(smt * 16 + r + hi * 8) * SP + snt * 16 + ln] = acc[r] * scale;
    }
    __syncthreads();

    // ---- Phase 3: masked row softmax into f16 attn (padded rows/cols -> 0)
    if (tid < SP) {
        const int row = tid;
        if (row < SS) {
            float m = -1e30f;
            for (int c = 0; c < SS; ++c) m = fmaxf(m, u.s.Sc[row * SP + c]);
            float sum = 0.0f;
            for (int c = 0; c < SS; ++c) sum += __expf(u.s.Sc[row * SP + c] - m);
            const float inv = 1.0f / sum;
            for (int c = 0; c < SS; ++c)
                u.s.At[row * SP + c] = (f16)(__expf(u.s.Sc[row * SP + c] - m) * inv);
            for (int c = SS; c < SP; ++c) u.s.At[row * SP + c] = (f16)0.0f;
        } else {
            for (int c = 0; c < SP; ++c) u.s.At[row * SP + c] = (f16)0.0f;
        }
    }
    __syncthreads();

    // ---- Phase 4: ctx = attn @ Vc  (K = 64 -> two k-steps)
    {
        v8f acc = {};
        #pragma unroll
        for (int ks = 0; ks < 2; ++ks) {
            v16h a  = load_a(u.s.At, mt * 16, ks * 32, SP, lane);
            v16h bf = load_bT(VcT_s, nt * 16, ks * 32, SP, lane);
            acc = wmma_f16(a, bf, acc);
        }
        #pragma unroll
        for (int r = 0; r < 8; ++r) {
            const int s  = mt * 16 + r + hi * 8;
            const int dc = nt * 16 + ln;
            if (s < SS && dc < DK)
                ctx[((size_t)(b * SS + s)) * D + h * DK + dc] = acc[r];
        }
    }
}

// ---------------------------------------------------------------------------
// Kernel 4: tq[m] = sum_q q_add[q] * tanh( ctx[m,:] @ W_add[:,q] + b_add[q] )
// 16 ctx rows per block; WMMA over packed W_add fragments; ds_add_f32 reduce.
// ---------------------------------------------------------------------------
__global__ __launch_bounds__(256) void pool_kernel(
    const float* __restrict__ ctx, const f16* __restrict__ Apk,
    const float* __restrict__ qadd, float* __restrict__ tq)
{
    __shared__ __align__(64) f16 Ar[16 * KP];
    __shared__ float rowsum[16];

    const int tid = threadIdx.x, wave = tid >> 5, lane = tid & 31;
    const int hi = lane >> 4, ln = lane & 15;
    const int m0 = blockIdx.x * 16;

    if (tid < 16) rowsum[tid] = 0.0f;
    for (int idx = tid; idx < 16 * KP; idx += 256) {
        const int r = idx / KP, c = idx % KP;
        float v = (c < D) ? ctx[(size_t)(m0 + r) * D + c]
                          : (c == D ? 1.0f : 0.0f);      // bias column
        Ar[idx] = (f16)v;
    }
    __syncthreads();

    for (int nt = wave; nt < 13; nt += 8) {
        v8f acc = {};
        #pragma unroll
        for (int kt = 0; kt < 10; ++kt) {
            v16h a  = load_a(Ar, 0, kt * 32, KP, lane);
            v16h bf = *(const v16h*)(Apk + (size_t)(kt * 13 + nt) * 512 + lane * 16);
            acc = wmma_f16(a, bf, acc);
        }
        const int col = nt * 16 + ln;
        const float q = (col < QDIM) ? qadd[col] : 0.0f;
        #pragma unroll
        for (int r = 0; r < 8; ++r) {
            const float val = tanhf(acc[r]) * q;
            atomicAdd(&rowsum[r + hi * 8], val);         // ds_add_f32
        }
    }
    __syncthreads();
    if (tid < 16) tq[m0 + tid] = rowsum[tid];
}

// ---------------------------------------------------------------------------
// Kernel 5: per-batch softmax over tq and weighted sum of ctx rows.
// ---------------------------------------------------------------------------
__global__ __launch_bounds__(256) void final_kernel(
    const float* __restrict__ tq, const float* __restrict__ ctx,
    float* __restrict__ out)
{
    __shared__ float a_s[SS];
    __shared__ float red[2];
    const int b = blockIdx.x, tid = threadIdx.x;

    if (tid == 0) {
        float m = -1e30f;
        for (int s = 0; s < SS; ++s) m = fmaxf(m, tq[b * SS + s]);
        float sum = 0.0f;
        for (int s = 0; s < SS; ++s) sum += __expf(tq[b * SS + s] - m);
        red[0] = m; red[1] = 1.0f / sum;
    }
    __syncthreads();
    if (tid < SS) a_s[tid] = __expf(tq[b * SS + tid] - red[0]) * red[1];
    __syncthreads();

    for (int d = tid; d < D; d += 256) {
        float acc = 0.0f;
        for (int s = 0; s < SS; ++s)
            acc += a_s[s] * ctx[((size_t)(b * SS + s)) * D + d];
        out[(size_t)b * D + d] = acc;
    }
}

// ---------------------------------------------------------------------------
// Launcher
// ---------------------------------------------------------------------------
extern "C" void kernel_launch(void* const* d_in, const int* in_sizes, int n_in,
                              void* d_out, int out_size, void* d_ws, size_t ws_size,
                              hipStream_t stream)
{
    (void)in_sizes; (void)n_in; (void)out_size; (void)ws_size;

    const int*   ids = (const int*)  d_in[0];
    const float* uv  = (const float*)d_in[1];
    const float* emb = (const float*)d_in[2];
    const float* Wq  = (const float*)d_in[3];  const float* bq  = (const float*)d_in[4];
    const float* Wk  = (const float*)d_in[5];  const float* bk  = (const float*)d_in[6];
    const float* Wv  = (const float*)d_in[7];  const float* bv  = (const float*)d_in[8];
    const float* Wqp = (const float*)d_in[9];  const float* bqp = (const float*)d_in[10];
    const float* Wkp = (const float*)d_in[11]; const float* bkp = (const float*)d_in[12];
    const float* Wad = (const float*)d_in[13]; const float* bad = (const float*)d_in[14];
    const float* qad = (const float*)d_in[15];
    float* out = (float*)d_out;

    char* ws = (char*)d_ws;
    size_t off = 0;
    auto alloc = [&](size_t bytes) {
        size_t o = off;
        off = (off + bytes + 511) & ~(size_t)511;
        return o;
    };
    f16*   Xv   = (f16*)  (ws + alloc((size_t)ROWS * KP * sizeof(f16)));   // 32.8 MB
    f16*   Xe   = (f16*)  (ws + alloc((size_t)ROWS * KP * sizeof(f16)));   // 32.8 MB
    f16*   Wpk  = (f16*)  (ws + alloc((size_t)1500 * 512 * sizeof(f16)));  // 1.5 MB
    f16*   Apk  = (f16*)  (ws + alloc((size_t)130 * 512 * sizeof(f16)));   // 0.13 MB
    float* ctx  = (float*)(ws + alloc((size_t)ROWS * D * sizeof(float)));  // 61.4 MB
    float* tq   = (float*)(ws + alloc((size_t)ROWS * sizeof(float)));      // 0.2 MB

    pack_kernel<<<1630, 32, 0, stream>>>(Wq, Wk, Wv, Wqp, Wkp,
                                         bq, bk, bv, bqp, bkp,
                                         Wad, bad, Wpk, Apk);

    {
        const size_t total = 2ull * ROWS * (KP / 8);   // b128 chunks
        const int blocks = (int)((total + 255) / 256);
        prep_kernel<<<blocks, 256, 0, stream>>>(uv, emb, ids, Xv, Xe);
    }

    attn_kernel<<<BATCH * H, 256, 0, stream>>>(Xv, Xe, Wpk, ctx);

    pool_kernel<<<ROWS / 16, 256, 0, stream>>>(ctx, Apk, qad, tq);

    final_kernel<<<BATCH, 256, 0, stream>>>(tq, ctx, out);
}
